// MORE_89249420411436
// MI455X (gfx1250) — matrix operand order repaired and checked by reading, not compiled, not run on hardware
//
#include <hip/hip_runtime.h>

// ---------------------------------------------------------------------------
// 3-layer GCN on gfx1250 (MI455X), wave32 + WMMA.
//  - W matrices pre-packed once into per-lane f16 WMMA B-fragment layout
//    (32 contiguous bytes/lane/fragment -> one b128-pair load per fragment).
//  - GEMM: one wave per 16-row tile, NT=N/16 accumulators, 8 (or 4) v_wmma
//    per 32-wide k-step; A (the 102 MB HBM stream) is loaded exactly once.
//  - Graph aggregation: per-edge float4 gather + f32 atomics; h/agg buffers
//    (25.6 MB) are L2-resident, so atomics resolve on-chip.
// Dims: N=50000 (=16*3125), K_feat=512, hidden=128, out=64, E=1,600,000.
// ---------------------------------------------------------------------------

typedef __attribute__((ext_vector_type(16))) _Float16 v16h;
typedef __attribute__((ext_vector_type(8)))  float    v8f;

// ---------------- degree / normalization --------------------------------

__global__ void init_one_kernel(float* __restrict__ p, int n) {
  int i = blockIdx.x * blockDim.x + threadIdx.x;
  if (i < n) p[i] = 1.0f;   // +1 self-loop baked in
}

__global__ void deg_count_kernel(const long long* __restrict__ dst,
                                 float* __restrict__ deg, long long E) {
  long long e = (long long)blockIdx.x * blockDim.x + threadIdx.x;
  if (e < E) atomicAdd(&deg[(int)dst[e]], 1.0f);
}

__global__ void rsqrt_ip_kernel(float* __restrict__ p, int n) {
  int i = blockIdx.x * blockDim.x + threadIdx.x;
  if (i < n) p[i] = rsqrtf(p[i]);   // deg -> dinv in place
}

// ---------------- W pre-pack into WMMA B-fragment layout ----------------
// Fragment (kb, nt): lane l holds column n = nt*16 + (l&15), K values
// kb*32 + 16*(l>>4) + {0..15}, stored as 16 contiguous f16 (32 B).
// Wp index = ((kb*NT + nt)*32 + lane)*16 + v.

template<int K, int N>
__global__ void pack_w_kernel(const float* __restrict__ W,
                              _Float16* __restrict__ Wp) {
  constexpr int NT = N / 16;
  int tid = blockIdx.x * blockDim.x + threadIdx.x;
  if (tid >= K * N) return;
  int v    = tid & 15;
  int lane = (tid >> 4) & 31;
  int frag = tid >> 9;           // kb*NT + nt
  int nt   = frag % NT;
  int kb   = frag / NT;
  int half = lane >> 4;
  int n    = nt * 16 + (lane & 15);
  int k    = kb * 32 + 16 * half + v;
  Wp[tid] = (_Float16)W[k * N + n];
}

// ---------------- WMMA GEMM: H[M,N] = A[M,K] @ B[K,N] -------------------
// One wave per 16-row tile; NT accumulators cover all N columns. Per
// 32-wide k-step: one A fragment (4x float4 + cvt), NT packed-B fragment
// loads (32 B each), NT back-to-back v_wmma_f32_16x16x32_f16.

template<int K, int N>
__global__ void gemm_wmma_kernel(const float* __restrict__ A,
                                 const _Float16* __restrict__ Bp,
                                 float* __restrict__ H, int mtiles) {
  constexpr int NT = N / 16;
  constexpr int KB = K / 32;
  const int lane   = threadIdx.x & 31;
  const int wave   = threadIdx.x >> 5;
  const int lane16 = lane & 15;
  const int half   = lane >> 4;
  const int tile   = blockIdx.x * (blockDim.x >> 5) + wave;
  if (tile >= mtiles) return;   // whole wave exits; EXEC all-ones otherwise

  const long long m0  = (long long)tile * 16;
  const float*    arp = A + (m0 + lane16) * K;

  v8f acc[NT];
#pragma unroll
  for (int nt = 0; nt < NT; ++nt) acc[nt] = (v8f){};

  for (int kb = 0; kb < KB; ++kb) {
    // A fragment (16x32 f16): lane half selects K offset +0/+16; VGPRs 0..7
    // hold K[koff..koff+15] contiguously (matches packed B).
    const float* ap = arp + kb * 32 + 16 * half;
    float4 a0 = *(const float4*)(ap + 0);
    float4 a1 = *(const float4*)(ap + 4);
    float4 a2 = *(const float4*)(ap + 8);
    float4 a3 = *(const float4*)(ap + 12);
    v16h a;
    a[0]=(_Float16)a0.x;  a[1]=(_Float16)a0.y;  a[2]=(_Float16)a0.z;  a[3]=(_Float16)a0.w;
    a[4]=(_Float16)a1.x;  a[5]=(_Float16)a1.y;  a[6]=(_Float16)a1.z;  a[7]=(_Float16)a1.w;
    a[8]=(_Float16)a2.x;  a[9]=(_Float16)a2.y;  a[10]=(_Float16)a2.z; a[11]=(_Float16)a2.w;
    a[12]=(_Float16)a3.x; a[13]=(_Float16)a3.y; a[14]=(_Float16)a3.z; a[15]=(_Float16)a3.w;

    const v16h* bk = (const v16h*)Bp + (long long)kb * NT * 32 + lane;
#pragma unroll
    for (int nt = 0; nt < NT; ++nt) {
      v16h b = bk[nt * 32];     // one 32-byte packed fragment load
      acc[nt] = __builtin_amdgcn_wmma_f32_16x16x32_f16(
          /*neg_a=*/false, a, /*neg_b=*/false, b,
          /*c_mod=*/(short)0, acc[nt], /*reuse_a=*/false, /*reuse_b=*/false);
    }
  }

  // D (16x16 f32): VGPR v -> row v (lanes 0-15) / row v+8 (lanes 16-31).
#pragma unroll
  for (int nt = 0; nt < NT; ++nt)
#pragma unroll
    for (int v = 0; v < 8; ++v)
      H[(m0 + v + 8 * half) * N + nt * 16 + lane16] = acc[nt][v];
}

// ---------------- self-loop + bias init ---------------------------------
// out[n,c] = h[n,c] * dinv[n]^2 + b[c]   (fully initializes the agg buffer)

template<int C>
__global__ void self_bias_kernel(const float* __restrict__ h,
                                 const float* __restrict__ dinv,
                                 const float* __restrict__ b,
                                 float* __restrict__ out, int n) {
  long long gid = (long long)blockIdx.x * blockDim.x + threadIdx.x;
  if (gid >= (long long)n * C) return;
  int node = (int)(gid / C);
  int c    = (int)(gid % C);
  float di = dinv[node];
  out[gid] = h[gid] * di * di + b[c];
}

// ---------------- per-edge gather/scale/scatter-add ---------------------

template<int C>
__global__ void edge_scatter_kernel(const long long* __restrict__ src,
                                    const long long* __restrict__ dst,
                                    const float* __restrict__ dinv,
                                    const float* __restrict__ h,
                                    float* __restrict__ agg, long long E) {
  constexpr int RP = C / 4;
  long long gid = (long long)blockIdx.x * blockDim.x + threadIdx.x;
  if (gid >= E * RP) return;
  long long e = gid / RP;
  int       r = (int)(gid % RP);
  int s = (int)src[e];
  int d = (int)dst[e];
  float w = dinv[s] * dinv[d];
  float4 hv = *(const float4*)(h + (long long)s * C + 4 * r);
  float* o  = agg + (long long)d * C + 4 * r;
  atomicAdd(o + 0, hv.x * w);
  atomicAdd(o + 1, hv.y * w);
  atomicAdd(o + 2, hv.z * w);
  atomicAdd(o + 3, hv.w * w);
}

// ---------------- elementwise -------------------------------------------

__global__ void relu_ip_kernel(float* __restrict__ x, long long n) {
  long long i = (long long)blockIdx.x * blockDim.x + threadIdx.x;
  if (i < n) x[i] = fmaxf(x[i], 0.0f);
}

__global__ void add_relu_kernel(float* __restrict__ x,
                                const float* __restrict__ y, long long n) {
  long long i = (long long)blockIdx.x * blockDim.x + threadIdx.x;
  if (i < n) x[i] += fmaxf(y[i], 0.0f);
}

// ---------------- launch ------------------------------------------------

extern "C" void kernel_launch(void* const* d_in, const int* in_sizes, int n_in,
                              void* d_out, int out_size, void* d_ws, size_t ws_size,
                              hipStream_t stream) {
  const float*     feat  = (const float*)d_in[0];     // [N,512]
  const float*     struc = (const float*)d_in[1];     // [N,128]
  const long long* eidx  = (const long long*)d_in[2]; // [2,E] int64
  const float*     W1    = (const float*)d_in[3];     // [512,128]
  const float*     b1    = (const float*)d_in[4];
  const float*     W2    = (const float*)d_in[5];     // [128,128]
  const float*     b2    = (const float*)d_in[6];
  const float*     W3    = (const float*)d_in[7];     // [128,64]
  const float*     b3    = (const float*)d_in[8];
  float*           out   = (float*)d_out;             // [N,64]

  const int       NN = in_sizes[0] / 512;             // 50000 (16*3125)
  const long long E  = (long long)in_sizes[2] / 2;    // 1,600,000
  const long long* esrc = eidx;
  const long long* edst = eidx + E;

  // Workspace (floats): dinv | h | x | y | packed W1/W2/W3 (f16)  (~77 MB)
  float* ws   = (float*)d_ws;
  size_t npad = ((size_t)NN + 63) & ~(size_t)63;
  float* dinv = ws;
  float* h    = ws + npad;
  float* x    = h + (size_t)NN * 128;
  float* y    = x + (size_t)NN * 128;
  _Float16* w1p = (_Float16*)(y + (size_t)NN * 128);
  _Float16* w2p = w1p + 512 * 128;
  _Float16* w3p = w2p + 128 * 128;

  auto nb = [](long long n) { return (unsigned)((n + 255) / 256); };
  const int      mtiles = NN / 16;                   // 3125
  const unsigned gblk   = (unsigned)((mtiles + 7) / 8);

  // --- pack weights into WMMA fragment layout (tiny, L2-resident) ---
  pack_w_kernel<512, 128><<<nb(512 * 128), 256, 0, stream>>>(W1, w1p);
  pack_w_kernel<128, 128><<<nb(128 * 128), 256, 0, stream>>>(W2, w2p);
  pack_w_kernel<128,  64><<<nb(128 *  64), 256, 0, stream>>>(W3, w3p);

  // --- normalization: deg -> dinv ---
  init_one_kernel<<<nb(NN), 256, 0, stream>>>(dinv, NN);
  deg_count_kernel<<<nb(E), 256, 0, stream>>>(edst, dinv, E);
  rsqrt_ip_kernel<<<nb(NN), 256, 0, stream>>>(dinv, NN);

  // --- conv1: feat[N,512] @ W1 -> h; aggregate -> x; relu ---
  gemm_wmma_kernel<512, 128><<<gblk, 256, 0, stream>>>(feat, w1p, h, mtiles);
  self_bias_kernel<128><<<nb((long long)NN * 128), 256, 0, stream>>>(h, dinv, b1, x, NN);
  edge_scatter_kernel<128><<<nb(E * 32), 256, 0, stream>>>(esrc, edst, dinv, h, x, E);
  relu_ip_kernel<<<nb((long long)NN * 128), 256, 0, stream>>>(x, (long long)NN * 128);

  // --- conv2: struc[N,128] @ W2 -> h; aggregate -> y; x += relu(y) ---
  gemm_wmma_kernel<128, 128><<<gblk, 256, 0, stream>>>(struc, w2p, h, mtiles);
  self_bias_kernel<128><<<nb((long long)NN * 128), 256, 0, stream>>>(h, dinv, b2, y, NN);
  edge_scatter_kernel<128><<<nb(E * 32), 256, 0, stream>>>(esrc, edst, dinv, h, y, E);
  add_relu_kernel<<<nb((long long)NN * 128), 256, 0, stream>>>(x, y, (long long)NN * 128);

  // --- conv3: x[N,128] @ W3 -> h[N,64]; aggregate -> out ---
  gemm_wmma_kernel<128, 64><<<gblk, 256, 0, stream>>>(x, w3p, h, mtiles);
  self_bias_kernel<64><<<nb((long long)NN * 64), 256, 0, stream>>>(h, dinv, b3, out, NN);
  edge_scatter_kernel<64><<<nb(E * 16), 256, 0, stream>>>(esrc, edst, dinv, h, out, E);
}
